// VQVAE_55190329753912
// MI455X (gfx1250) — matrix operand order repaired and checked by reading, not compiled, not run on hardware
//
#include <hip/hip_runtime.h>

// ---------------------------------------------------------------------------
// VQ-VAE forward for MI455X (gfx1250), wave32 + WMMA.
//  - all channel contractions (e2,e3,d1,d2, VQ z@emb^T) use v_wmma_f32_16x16x32_f16
//  - intermediates stored f16 (memory-bound workload, 23.3 TB/s HBM)
//  - transpose convs = flipped-weight convs with lhs-dilation handled in im2col
//  - conv K-loop processes 64 K per barrier pair (2 WMMA / sync) to cut
//    barrier overhead in half vs the 32-K version
// ---------------------------------------------------------------------------

typedef __attribute__((ext_vector_type(16))) _Float16     v16h;
typedef __attribute__((ext_vector_type(8)))  float        v8f;
typedef __attribute__((ext_vector_type(4)))  unsigned int u32x4;

union Frag16 {
    u32x4        q[2];
    unsigned int u[8];
    v16h         h;
};

union H2U { _Float16 f; unsigned short u; };

__device__ __forceinline__ v8f wmma_f32_f16(v16h a, v16h b, v8f c) {
    // D = A(16x32 f16) * B(32x16 f16) + C(16x16 f32)
    return __builtin_amdgcn_wmma_f32_16x16x32_f16(
        /*neg_a=*/false, a, /*neg_b=*/false, b,
        /*c_mod=*/(short)0, c, /*reuse_a=*/false, /*reuse_b=*/false);
}

// ---------------------------------------------------------------------------
// Weight packer: repack OIHW (optionally flipped/transposed ConvTranspose
// weights, src layout (Cin,Cout,kh,kw)) into the exact per-lane A-fragment
// layout of V_WMMA_F32_16X16X32_F16:
//   lane l: M = l%16, h = l/16 ; dword v: K = h*8 + (v%4)*2 + (v/4)*16 (+1)
// dst dword index = ((ks32*OCT + octile)*32 + lane)*8 + v   (ks32 = K/32 block)
// ---------------------------------------------------------------------------
__global__ __launch_bounds__(256) void pack_w(
    const float* __restrict__ src, unsigned int* __restrict__ dst,
    int OC, int IC, int KH, int KW, int flip, int total)
{
    int d = blockIdx.x * 256 + threadIdx.x;
    if (d >= total) return;
    const int KHW  = KH * KW;
    const int KTOT = IC * KHW;
    const int OCT  = OC >> 4;
    int v      = d & 7;
    int lane   = (d >> 3) & 31;
    int octile = (d >> 8) % OCT;
    int ks     = (d >> 8) / OCT;
    int m = lane & 15, h = lane >> 4;
    int oc = octile * 16 + m;
    int k0 = ks * 32 + h * 8 + (v & 3) * 2 + ((v >> 2) << 4);
    unsigned int out = 0;
    for (int u = 0; u < 2; ++u) {
        int k = k0 + u;
        float w = 0.f;
        if (k < KTOT) {
            int ic = k / KHW;
            int r  = k - ic * KHW;
            int kh = r / KW;
            int kw = r - kh * KW;
            if (flip) w = src[(((size_t)ic * OC + oc) * KH + (KH - 1 - kh)) * KW + (KW - 1 - kw)];
            else      w = src[(((size_t)oc * IC + ic) * KH + kh) * KW + kw];
        }
        H2U cv; cv.f = (_Float16)w;
        out |= (unsigned int)cv.u << (u * 16);
    }
    dst[d] = out;
}

// codebook: f16 copy + ||e||^2 per code
__global__ __launch_bounds__(256) void prep_emb(
    const float* __restrict__ emb, _Float16* __restrict__ embh, float* __restrict__ e2n)
{
    int code = blockIdx.x * 256 + threadIdx.x;
    if (code >= 512) return;
    float s = 0.f;
    for (int c = 0; c < 64; ++c) {
        float v = emb[code * 64 + c];
        s += v * v;
        embh[code * 64 + c] = (_Float16)v;
    }
    e2n[code] = s;
}

// ---------------------------------------------------------------------------
// e1: Conv 1->32, 4x4, s2, p1, ReLU.  Direct (IC=1 is a degenerate GEMM).
// x (64,1,256,256) f32 -> h1 (64,32,128,128) f16
// ---------------------------------------------------------------------------
__global__ __launch_bounds__(256) void e1_conv(
    const float* __restrict__ x, const float* __restrict__ w,
    const float* __restrict__ b, _Float16* __restrict__ y)
{
    __shared__ float swt[512];
    __shared__ float sb[32];
    for (int i = threadIdx.x; i < 512; i += 256) swt[i] = w[i];
    if (threadIdx.x < 32) sb[threadIdx.x] = b[threadIdx.x];
    __syncthreads();
    int px = blockIdx.x * 256 + threadIdx.x;         // 64*128*128 total
    int n = px >> 14, s = px & 16383;
    int oh = s >> 7, ow = s & 127;
    float patch[16];
#pragma unroll
    for (int kh = 0; kh < 4; ++kh)
#pragma unroll
        for (int kw = 0; kw < 4; ++kw) {
            int ih = oh * 2 - 1 + kh, iw = ow * 2 - 1 + kw;
            float v = 0.f;
            if (ih >= 0 && ih < 256 && iw >= 0 && iw < 256)
                v = x[((size_t)n << 16) + (ih << 8) + iw];
            patch[kh * 4 + kw] = v;
        }
#pragma unroll 4
    for (int oc = 0; oc < 32; ++oc) {
        float a = sb[oc];
#pragma unroll
        for (int t = 0; t < 16; ++t) a = fmaf(swt[oc * 16 + t], patch[t], a);
        y[(((size_t)n * 32 + oc) << 14) + s] = (_Float16)fmaxf(a, 0.f);
    }
}

// ---------------------------------------------------------------------------
// Generic implicit-GEMM conv via WMMA.
//  GEMM: M = OC (A = packed weights), N = pixels (B = im2col), K = IC*KH*KW
//  Workgroup: 256 threads = 8 waves = OCT oc-tiles x (8/OCT) pixel-tiles.
//  K processed 64 per iteration: stage 64 K to LDS, then 2 WMMA per wave.
//  TCONV: lhs-dilation-2 conv (PAD = k-1-p), zeros gathered at odd coords.
// ---------------------------------------------------------------------------
template<int IC, int OC, int KH, int KW, int STRIDE, int PAD,
         int HIN, int WIN, int HOUT, int WOUT,
         bool TCONV, bool RELU, bool INH, bool OUTH>
__global__ __launch_bounds__(256) void conv_wmma(
    const void* __restrict__ xin, const unsigned int* __restrict__ wpk,
    const float* __restrict__ bias, void* __restrict__ yout)
{
    constexpr int KHW    = KH * KW;
    constexpr int KTOT   = IC * KHW;
    constexpr int KSTEPS = KTOT / 64;           // 64 K per iteration
    constexpr int OCT    = OC / 16;
    constexpr int PXT    = 8 / OCT;
    constexpr int PXWG   = PXT * 16;
    constexpr int HWO    = HOUT * WOUT;
    constexpr int LROW   = 72;                  // f16 row stride (16B-aligned rows)
    static_assert(KTOT % 64 == 0, "K must be multiple of 64");

    __shared__ __align__(16) _Float16 sp[PXWG * LROW];

    const int tid    = threadIdx.x;
    const int wave   = tid >> 5;
    const int lane   = tid & 31;
    const int m      = lane & 15;
    const int h      = lane >> 4;
    const int octile = wave % OCT;
    const int pxtile = wave / OCT;
    const int pxbase = blockIdx.x * PXWG;

    const float*    xf = (const float*)xin;
    const _Float16* xh = (const _Float16*)xin;

    v8f acc = {0.f, 0.f, 0.f, 0.f, 0.f, 0.f, 0.f, 0.f};

    for (int ks = 0; ks < KSTEPS; ++ks) {
        __syncthreads();
        // stage im2col patch (PXWG pixels x 64 K) as f16 into LDS
        for (int e = tid; e < PXWG * 64; e += 256) {
            int pl = e >> 6;
            int kl = e & 63;
            int k  = ks * 64 + kl;
            int ic = k / KHW;
            int r  = k - ic * KHW;
            int kh = r / KW;
            int kw = r - kh * KW;
            int px = pxbase + pl;
            int n  = px / HWO;
            int s  = px - n * HWO;
            int oh = s / WOUT;
            int ow = s - oh * WOUT;
            _Float16 val = (_Float16)0.f;
            if (!TCONV) {
                int ih = oh * STRIDE - PAD + kh;
                int iw = ow * STRIDE - PAD + kw;
                if (ih >= 0 && ih < HIN && iw >= 0 && iw < WIN) {
                    size_t idx = ((size_t)(n * IC + ic) * HIN + ih) * WIN + iw;
                    val = INH ? xh[idx] : (_Float16)xf[idx];
                }
            } else {
                int ih2 = oh - PAD + kh;
                int iw2 = ow - PAD + kw;
                if (ih2 >= 0 && iw2 >= 0 && !(ih2 & 1) && !(iw2 & 1)) {
                    int ih = ih2 >> 1, iw = iw2 >> 1;
                    if (ih < HIN && iw < WIN) {
                        size_t idx = ((size_t)(n * IC + ic) * HIN + ih) * WIN + iw;
                        val = INH ? xh[idx] : (_Float16)xf[idx];
                    }
                }
            }
            sp[pl * LROW + kl] = val;
        }
        __syncthreads();

        // two consecutive 32-K packed blocks -> two WMMA per barrier pair
        Frag16 a0, a1, b0, b1;
        const u32x4* pw0 = (const u32x4*)wpk + ((size_t)((ks * 2 + 0) * OCT + octile) * 32 + lane) * 2;
        const u32x4* pw1 = (const u32x4*)wpk + ((size_t)((ks * 2 + 1) * OCT + octile) * 32 + lane) * 2;
        a0.q[0] = pw0[0]; a0.q[1] = pw0[1];
        a1.q[0] = pw1[0]; a1.q[1] = pw1[1];
        if (ks + 1 < KSTEPS)
            __builtin_prefetch((const void*)(pw0 + (size_t)OCT * 128), 0, 0);
        const u32x4* pb0 = (const u32x4*)(sp + (pxtile * 16 + m) * LROW + h * 8);
        const u32x4* pb1 = (const u32x4*)(sp + (pxtile * 16 + m) * LROW + 32 + h * 8);
        b0.q[0] = pb0[0]; b0.q[1] = pb0[2];
        b1.q[0] = pb1[0]; b1.q[1] = pb1[2];
        acc = wmma_f32_f16(a0.h, b0.h, acc);
        acc = wmma_f32_f16(a1.h, b1.h, acc);
    }

    // D layout: lane col N = l%16 = pixel, VGPR v row M = h*8+v = oc-in-tile
    {
        int px = pxbase + pxtile * 16 + m;
        int n = px / HWO;
        int s = px - n * HWO;
        float*    yf = (float*)yout;
        _Float16* yh = (_Float16*)yout;
#pragma unroll
        for (int v = 0; v < 8; ++v) {
            int   oc = octile * 16 + h * 8 + v;
            float r  = acc[v] + bias[oc];
            if (RELU) r = fmaxf(r, 0.f);
            size_t idx = (size_t)(n * OC + oc) * HWO + s;
            if (OUTH) yh[idx] = (_Float16)r;
            else      yf[idx] = r;
        }
    }
}

// ---------------------------------------------------------------------------
// VQ: per pixel argmin_code(||e||^2 - 2 z.e)   (||z||^2 constant -> dropped)
// One wave = 16 pixels x all 512 codes via WMMA, then 16-lane argmin shuffle.
// ze (d_out slice, f32) -> zq (d_out slice, f32)
// ---------------------------------------------------------------------------
__global__ __launch_bounds__(256) void vq_kernel(
    const float* __restrict__ ze, const _Float16* __restrict__ embh,
    const float* __restrict__ e2n, const float* __restrict__ embf,
    float* __restrict__ zq)
{
    __shared__ __align__(16) _Float16 sp[128 * 68];
    __shared__ int sidx[8][16];
    const int tid    = threadIdx.x;
    const int pxbase = blockIdx.x * 128;

    // stage 128 pixels x 64 channels of ze as f16 (coalesced over spatial)
    for (int e = tid; e < 128 * 64; e += 256) {
        int c  = e >> 7;
        int pl = e & 127;
        int px = pxbase + pl;
        int n = px >> 12, s = px & 4095;
        sp[pl * 68 + c] = (_Float16)ze[(((size_t)n * 64 + c) << 12) + s];
    }
    __syncthreads();

    const int wave = tid >> 5, lane = tid & 31;
    const int m = lane & 15, h = lane >> 4;
    const int prow = wave * 16 + m;

    // A fragments: z rows (M=pixel, K=channel), two K-steps of 32
    Frag16 a0, a1;
#pragma unroll
    for (int v = 0; v < 8; ++v) {
        int c = h * 8 + (v & 3) * 2 + ((v >> 2) << 4);
        a0.u[v] = *(const unsigned int*)(sp + prow * 68 + c);
        a1.u[v] = *(const unsigned int*)(sp + prow * 68 + 32 + c);
    }

    float best[8];
    int   bidx[8];
#pragma unroll
    for (int v = 0; v < 8; ++v) { best[v] = 3.0e38f; bidx[v] = 0; }

    for (int cb = 0; cb < 32; ++cb) {
        int code = cb * 16 + m;                       // this lane's column
        Frag16 b0, b1;
        const u32x4* pe0 = (const u32x4*)(embh + code * 64 + h * 8);
        const u32x4* pe1 = (const u32x4*)(embh + code * 64 + 32 + h * 8);
        b0.q[0] = pe0[0]; b0.q[1] = pe0[2];
        b1.q[0] = pe1[0]; b1.q[1] = pe1[2];
        v8f acc = {0.f, 0.f, 0.f, 0.f, 0.f, 0.f, 0.f, 0.f};
        acc = wmma_f32_f16(a0.h, b0.h, acc);
        acc = wmma_f32_f16(a1.h, b1.h, acc);
        float en = e2n[code];
#pragma unroll
        for (int v = 0; v < 8; ++v) {
            float d = fmaf(-2.f, acc[v], en);
            if (d < best[v]) { best[v] = d; bidx[v] = code; }  // strict < => lowest code wins
        }
    }

    // argmin across the 16 lanes of each half (rows M = h*8+v)
#pragma unroll
    for (int off = 1; off < 16; off <<= 1) {
#pragma unroll
        for (int v = 0; v < 8; ++v) {
            float od = __shfl_xor(best[v], off, 32);
            int   oi = __shfl_xor(bidx[v], off, 32);
            if (od < best[v] || (od == best[v] && oi < bidx[v])) { best[v] = od; bidx[v] = oi; }
        }
    }
#pragma unroll
    for (int v = 0; v < 8; ++v)
        if (m == v) sidx[wave][h * 8 + v] = bidx[v];
    __syncthreads();

    // gather zq = emb[idx] (f32 exact), coalesced over spatial
    int code = sidx[wave][m];
    int px = pxbase + wave * 16 + m;
    int n = px >> 12, s = px & 4095;
    const float* er = embf + (size_t)code * 64;
    for (int c = h * 32; c < h * 32 + 32; ++c)
        zq[(((size_t)n * 64 + c) << 12) + s] = er[c];
}

// ---------------------------------------------------------------------------
// d3: ConvTranspose 64->1, 4x4, s2, p1 (OC=1 degenerate GEMM -> direct).
// in (64,64,128,128) f16 (L2-resident, 134MB < 192MB L2) -> x_hat f32
// ---------------------------------------------------------------------------
__global__ __launch_bounds__(256) void d3_conv(
    const _Float16* __restrict__ x, const float* __restrict__ w,
    const float* __restrict__ b, float* __restrict__ y)
{
    __shared__ float swt[64 * 16];
    for (int i = threadIdx.x; i < 1024; i += 256) {
        int ic = i >> 4, t = i & 15;
        swt[i] = w[ic * 16 + 15 - t];   // spatial flip of (Cin=64,Cout=1,4,4)
    }
    __syncthreads();
    int px = blockIdx.x * 256 + threadIdx.x;        // 64*256*256 total
    int n = px >> 16, s = px & 65535;
    int oh = s >> 8, ow = s & 255;
    float acc = b[0];
#pragma unroll
    for (int kh = 0; kh < 4; ++kh) {
        int ih2 = oh - 2 + kh;
        if (ih2 < 0 || (ih2 & 1) || (ih2 >> 1) >= 128) continue;
        int ih = ih2 >> 1;
#pragma unroll
        for (int kw = 0; kw < 4; ++kw) {
            int iw2 = ow - 2 + kw;
            if (iw2 < 0 || (iw2 & 1) || (iw2 >> 1) >= 128) continue;
            int iw = iw2 >> 1;
            const _Float16* xp = x + (((size_t)n * 64) << 14) + (ih << 7) + iw;
            int widx = kh * 4 + kw;
            for (int ic = 0; ic < 64; ++ic)
                acc = fmaf(swt[ic * 16 + widx], (float)xp[(size_t)ic << 14], acc);
        }
    }
    y[px] = acc;
}

// ---------------------------------------------------------------------------
extern "C" void kernel_launch(void* const* d_in, const int* in_sizes, int n_in,
                              void* d_out, int out_size, void* d_ws, size_t ws_size,
                              hipStream_t stream)
{
    (void)in_sizes; (void)n_in;
    const float* x    = (const float*)d_in[0];
    const float* e1_w = (const float*)d_in[1];
    const float* e1_b = (const float*)d_in[2];
    const float* e2_w = (const float*)d_in[3];
    const float* e2_b = (const float*)d_in[4];
    const float* e3_w = (const float*)d_in[5];
    const float* e3_b = (const float*)d_in[6];
    const float* emb  = (const float*)d_in[7];
    const float* d1_w = (const float*)d_in[8];
    const float* d1_b = (const float*)d_in[9];
    const float* d2_w = (const float*)d_in[10];
    const float* d2_b = (const float*)d_in[11];
    const float* d3_w = (const float*)d_in[12];
    const float* d3_b = (const float*)d_in[13];

    // workspace layout (lifetimes overlapped): region A = h1 then d2out,
    // region B = h2 then d1out, then packed weights / codebook.
    char* ws = (char*)d_ws;
    constexpr size_t SZ_A     = (size_t)64 * 64 * 128 * 128 * 2;  // 134,217,728
    constexpr size_t SZ_B     = (size_t)64 * 64 * 64 * 64 * 2;    //  33,554,432
    constexpr size_t OFF_B    = SZ_A;
    constexpr size_t OFF_E2PK = OFF_B + SZ_B;
    constexpr size_t OFF_E3PK = OFF_E2PK + 65536;
    constexpr size_t OFF_D1PK = OFF_E3PK + 73728;
    constexpr size_t OFF_D2PK = OFF_D1PK + 36864;
    constexpr size_t OFF_EMBH = OFF_D2PK + 65536;
    constexpr size_t OFF_E2N  = OFF_EMBH + 65536;
    constexpr size_t WS_NEED  = OFF_E2N + 2048;
    if (ws_size < WS_NEED || out_size < 37748736) return;

    _Float16* h1  = (_Float16*)(ws);             // (64,32,128,128) f16
    _Float16* d2o = (_Float16*)(ws);             // (64,64,128,128) f16 (reuses A)
    _Float16* h2  = (_Float16*)(ws + OFF_B);     // (64,64,64,64) f16
    _Float16* d1o = (_Float16*)(ws + OFF_B);     // (64,32,64,64) f16 (reuses B)
    unsigned int* e2pk = (unsigned int*)(ws + OFF_E2PK);
    unsigned int* e3pk = (unsigned int*)(ws + OFF_E3PK);
    unsigned int* d1pk = (unsigned int*)(ws + OFF_D1PK);
    unsigned int* d2pk = (unsigned int*)(ws + OFF_D2PK);
    _Float16* embh = (_Float16*)(ws + OFF_EMBH);
    float*    e2n  = (float*)(ws + OFF_E2N);

    float* out  = (float*)d_out;                 // (x_hat, ze, zq) concatenated
    float* xhat = out;
    float* ze   = out + (size_t)64 * 256 * 256;  // +4,194,304
    float* zq   = ze  + (size_t)64 * 64 * 64 * 64;

    dim3 B(256);

    // ---- weight / codebook prep (tiny) ----
    pack_w<<<dim3(64), B, 0, stream>>>(e2_w, e2pk, 64, 32, 4, 4, 0, 16384);
    pack_w<<<dim3(72), B, 0, stream>>>(e3_w, e3pk, 64, 64, 3, 3, 0, 18432);
    pack_w<<<dim3(36), B, 0, stream>>>(d1_w, d1pk, 32, 64, 3, 3, 1, 9216);   // flipped tconv
    pack_w<<<dim3(64), B, 0, stream>>>(d2_w, d2pk, 64, 32, 4, 4, 1, 16384);  // flipped tconv
    prep_emb<<<dim3(2), B, 0, stream>>>(emb, embh, e2n);

    // ---- encoder ----
    e1_conv<<<dim3(4096), B, 0, stream>>>(x, e1_w, e1_b, h1);
    conv_wmma<32, 64, 4, 4, 2, 1, 128, 128, 64, 64, false, true, true, true>
        <<<dim3(8192), B, 0, stream>>>(h1, e2pk, e2_b, h2);
    conv_wmma<64, 64, 3, 3, 1, 1, 64, 64, 64, 64, false, true, true, false>
        <<<dim3(8192), B, 0, stream>>>(h2, e3pk, e3_b, ze);

    // ---- vector quantization (dec_in == zq in forward value) ----
    vq_kernel<<<dim3(2048), B, 0, stream>>>(ze, embh, e2n, emb, zq);

    // ---- decoder ----
    conv_wmma<64, 32, 3, 3, 1, 1, 64, 64, 64, 64, false, true, false, true>
        <<<dim3(4096), B, 0, stream>>>(zq, d1pk, d1_b, d1o);
    conv_wmma<32, 64, 4, 4, 1, 2, 64, 64, 128, 128, true, true, true, true>
        <<<dim3(32768), B, 0, stream>>>(d1o, d2pk, d2_b, d2o);
    d3_conv<<<dim3(16384), B, 0, stream>>>(d2o, d3_w, d3_b, xhat);
}
